// Seq2Seq_13726715478707
// MI455X (gfx1250) — compile-verified, matrix operand order checked by reading, not run on hardware
//
#include <hip/hip_runtime.h>
#include <hip/hip_bf16.h>

typedef _Float16 v16h __attribute__((ext_vector_type(16)));
typedef _Float16 v8h  __attribute__((ext_vector_type(8)));
typedef float    v8f  __attribute__((ext_vector_type(8)));

#define NB  32
#define NS  128
#define NT  64
#define NH  512
#define NV  32000

// Load a 16-half WMMA fragment slice: lane<16 holds K {0..7,16..23}, lane>=16 holds {8..15,24..31}.
// Caller passes p already offset by (lane>>4)*8 halves.
__device__ __forceinline__ v16h frag_ld(const _Float16* __restrict__ p) {
  v8h lo = *reinterpret_cast<const v8h*>(p);
  v8h hi = *reinterpret_cast<const v8h*>(p + 16);
  v16h r;
#pragma unroll
  for (int i = 0; i < 8; ++i) { r[i] = lo[i]; r[i + 8] = hi[i]; }
  return r;
}

__device__ __forceinline__ v8f wmma_f16(v16h a, v16h b, v8f c) {
  return __builtin_amdgcn_wmma_f32_16x16x32_f16(false, a, false, b, (short)0, c, false, false);
}

__device__ __forceinline__ float sigm(float x) { return 1.f / (1.f + expf(-x)); }

// ---- CDNA5 async DMA: global -> LDS, tracked with ASYNCcnt ----
__device__ __forceinline__ void async_ld_b128(unsigned lds_off, const _Float16* gaddr) {
  asm volatile("global_load_async_to_lds_b128 %0, %1, off"
               :: "v"(lds_off), "v"(gaddr) : "memory");
}
__device__ __forceinline__ void wait_async0() {
  asm volatile("s_wait_asynccnt 0" ::: "memory");
}

// ---------------- generic WMMA GEMM: C[M,N] = A[M,K] @ W[N,K]^T (+bias) ----------------
// grid = (N/128, Mpad/64), block = 256 (8 waves).
// Each wave: one 16-col N-tile x four 16-row M-tiles (B fragment reused 4x from regs).
// A strip [64 x K] is staged through LDS in double-buffered 128-wide K chunks using
// global_load_async_to_lds_b128 (shared by all 8 waves -> 8x less A traffic).
// A fragments are software-pipelined (next k-step preloaded into distinct regs) so the
// ds_load -> wmma chain doesn't serialize on register reuse.
// K % 128 == 0 required. Stores masked to rows < M (A allocations are padded).
__global__ void __launch_bounds__(256)
gemm_wmma(const _Float16* __restrict__ A, int lda,
          const _Float16* __restrict__ W, int ldw,
          const float* __restrict__ bias,
          float* __restrict__ C, int ldc, int N, int K, int M) {
  __shared__ __align__(16) _Float16 Ash[2][64 * 128];
  const int lane = threadIdx.x & 31;
  const int wv   = threadIdx.x >> 5;
  const int nt   = blockIdx.x * 8 + wv;
  const int rowBase = blockIdx.y * 64;
  const int col  = nt * 16 + (lane & 15);
  const int koff = (lane >> 4) * 8;
  const int arow = (lane & 15) * 128 + koff;     // A frag base within a 16-row tile

  float bv = bias ? bias[col] : 0.f;
  v8f acc[4];
#pragma unroll
  for (int m = 0; m < 4; ++m)
#pragma unroll
    for (int v = 0; v < 8; ++v) acc[m][v] = bv;

  const _Float16* wp = W + (size_t)col * ldw + koff;

  // stage chunk 0
#pragma unroll
  for (int it = 0; it < 4; ++it) {
    int idx = threadIdx.x + it * 256;            // 0..1023
    int row = idx >> 4, seg = idx & 15;          // 64 rows x 16 b128 segs
    async_ld_b128((unsigned)(uintptr_t)&Ash[0][row * 128 + seg * 8],
                  A + (size_t)(rowBase + row) * lda + seg * 8);
  }

  const int nc = K >> 7;                         // K / 128
  for (int c = 0; c < nc; ++c) {
    wait_async0();
    __syncthreads();
    if (c + 1 < nc) {
      const int kB = (c + 1) << 7;
      _Float16* dst = Ash[(c + 1) & 1];
#pragma unroll
      for (int it = 0; it < 4; ++it) {
        int idx = threadIdx.x + it * 256;
        int row = idx >> 4, seg = idx & 15;
        async_ld_b128((unsigned)(uintptr_t)&dst[row * 128 + seg * 8],
                      A + (size_t)(rowBase + row) * lda + kB + seg * 8);
      }
    }
    const _Float16* bufp = Ash[c & 1] + arow;
    // preload k-step 0 A fragments
    v16h af[4];
#pragma unroll
    for (int m = 0; m < 4; ++m) af[m] = frag_ld(bufp + m * 16 * 128);
#pragma unroll
    for (int kk = 0; kk < 128; kk += 32) {
      v16h bf = frag_ld(wp + (c << 7) + kk);
      v16h nf[4];
      if (kk + 32 < 128) {
#pragma unroll
        for (int m = 0; m < 4; ++m) nf[m] = frag_ld(bufp + m * 16 * 128 + kk + 32);
      }
#pragma unroll
      for (int m = 0; m < 4; ++m) acc[m] = wmma_f16(af[m], bf, acc[m]);
#pragma unroll
      for (int m = 0; m < 4; ++m) af[m] = nf[m];
    }
    __syncthreads();
  }

#pragma unroll
  for (int m = 0; m < 4; ++m) {
    int r0 = rowBase + m * 16 + (lane >> 4) * 8;
#pragma unroll
    for (int v = 0; v < 8; ++v)
      if (r0 + v < M) C[(size_t)(r0 + v) * ldc + col] = acc[m][v];
  }
}

// ---------------- encoder recurrence: one workgroup, loops over S steps ----------------
// wave w: mtile = w>>4, gate-column base jb = (w&15)*32 -> owns i/f/g/o fragments in regs.
__global__ void __launch_bounds__(1024)
enc_recurrent(const float* __restrict__ Gx,      // [S*B, 4H] precomputed x@Wih^T
              const _Float16* __restrict__ Whh,  // [4H, H] f16
              const float* __restrict__ bih, const float* __restrict__ bhh,
              _Float16* __restrict__ EncOut,     // [S*B, H] f16 (time-major)
              float* __restrict__ cf_out) {      // [B*H]
  __shared__ __align__(16) _Float16 hbuf[NB * NH];
  const int tid = threadIdx.x;
  const int lane = tid & 31, wv = tid >> 5;
  const int mt = wv >> 4, jb = (wv & 15) * 32;
  const int rr = mt * 16 + (lane & 15);
  const int koff = (lane >> 4) * 8;
  const int r0 = mt * 16 + (lane >> 4) * 8;
  const int cj = lane & 15;

  for (int i = tid; i < NB * NH; i += 1024) hbuf[i] = (_Float16)0.f;
  float c[2][8];
#pragma unroll
  for (int jt = 0; jt < 2; ++jt)
#pragma unroll
    for (int v = 0; v < 8; ++v) c[jt][v] = 0.f;
  __syncthreads();

  for (int s = 0; s < NS; ++s) {
    v8f acc[4][2];
#pragma unroll
    for (int g = 0; g < 4; ++g)
#pragma unroll
      for (int jt = 0; jt < 2; ++jt) {
        int ccol = g * NH + jb + jt * 16 + cj;
        float bsum = bih[ccol] + bhh[ccol];
#pragma unroll
        for (int v = 0; v < 8; ++v)
          acc[g][jt][v] = Gx[(size_t)(s * NB + r0 + v) * 2048 + ccol] + bsum;
      }
    const _Float16* hp = &hbuf[rr * NH + koff];
    for (int k = 0; k < NH; k += 32) {
      v16h af = frag_ld(hp + k);
#pragma unroll
      for (int g = 0; g < 4; ++g)
#pragma unroll
        for (int jt = 0; jt < 2; ++jt) {
          const _Float16* wp = Whh + (size_t)(g * NH + jb + jt * 16 + cj) * NH + koff + k;
          acc[g][jt] = wmma_f16(af, frag_ld(wp), acc[g][jt]);
        }
    }
    __syncthreads();
#pragma unroll
    for (int jt = 0; jt < 2; ++jt)
#pragma unroll
      for (int v = 0; v < 8; ++v) {
        float si = sigm(acc[0][jt][v]);
        float sf = sigm(acc[1][jt][v]);
        float gg = tanhf(acc[2][jt][v]);
        float so = sigm(acc[3][jt][v]);
        float cc = sf * c[jt][v] + si * gg;
        c[jt][v] = cc;
        float hh = so * tanhf(cc);
        int row = r0 + v, jcol = jb + jt * 16 + cj;
        hbuf[row * NH + jcol] = (_Float16)hh;
        EncOut[(size_t)(s * NB + row) * NH + jcol] = (_Float16)hh;
        if (s == NS - 1) cf_out[row * NH + jcol] = cc;
      }
    __syncthreads();
  }
}

// ---------------- decoder recurrence: one workgroup, loops over T-1 steps ----------------
__global__ void __launch_bounds__(1024)
dec_recurrent(const float* __restrict__ Gx,        // [63*B, 4H] embedding-side gates
              const _Float16* __restrict__ Wih,    // [4H, 2H] f16 (cols H.. = weighted part)
              const _Float16* __restrict__ Whh,    // [4H, H] f16
              const _Float16* __restrict__ Watt,   // [H, 2H] f16 (cols 0..H-1 = Wh)
              const float* __restrict__ bih, const float* __restrict__ bhh,
              const float* __restrict__ attn_v,
              const int* __restrict__ src,         // [B,S]
              const float* __restrict__ hid0,      // [B,H] tanh'd
              const float* __restrict__ cf0,       // [B,H]
              const _Float16* __restrict__ EncOut, // [S*B,H] f16
              const float* __restrict__ EncProj,   // [S*B,H] f32 (enc@We^T + attn_b)
              float* __restrict__ hW,              // [B,H] global scratch
              _Float16* __restrict__ Feat) {       // [63*B, 3H]  [h | weighted | e]
  __shared__ __align__(16) _Float16 hbuf[NB * NH];
  __shared__ __align__(16) _Float16 wbuf[NB * NH];
  const int tid = threadIdx.x;
  const int lane = tid & 31, wv = tid >> 5;
  const int mt = wv >> 4, jb = (wv & 15) * 32;
  const int rr = mt * 16 + (lane & 15);
  const int koff = (lane >> 4) * 8;
  const int r0 = mt * 16 + (lane >> 4) * 8;
  const int cj = lane & 15;

  for (int i = tid; i < NB * NH; i += 1024) hbuf[i] = (_Float16)hid0[i];
  float c[2][8];
#pragma unroll
  for (int jt = 0; jt < 2; ++jt)
#pragma unroll
    for (int v = 0; v < 8; ++v)
      c[jt][v] = cf0[(r0 + v) * NH + jb + jt * 16 + cj];
  __syncthreads();

  for (int t = 0; t < NT - 1; ++t) {
    // ---- Phase A: hW = h @ Wh^T  (M=32,N=512,K=512); wave -> 2 N-tiles ----
    {
      const int nt0 = (wv & 15), nt1 = (wv & 15) + 16;
      v8f a0, a1;
#pragma unroll
      for (int v = 0; v < 8; ++v) { a0[v] = 0.f; a1[v] = 0.f; }
      const _Float16* hp = &hbuf[rr * NH + koff];
      const _Float16* w0 = Watt + (size_t)(nt0 * 16 + cj) * 1024 + koff;
      const _Float16* w1 = Watt + (size_t)(nt1 * 16 + cj) * 1024 + koff;
      for (int k = 0; k < NH; k += 32) {
        v16h af = frag_ld(hp + k);
        a0 = wmma_f16(af, frag_ld(w0 + k), a0);
        a1 = wmma_f16(af, frag_ld(w1 + k), a1);
      }
#pragma unroll
      for (int v = 0; v < 8; ++v) {
        hW[(size_t)(r0 + v) * NH + nt0 * 16 + cj] = a0[v];
        hW[(size_t)(r0 + v) * NH + nt1 * 16 + cj] = a1[v];
      }
    }
    __threadfence_block();
    __syncthreads();

    // ---- Phase B: scores + softmax; wave wv owns batch row b = wv ----
    const int b = wv;
    float hv[16], vv[16];
#pragma unroll
    for (int e = 0; e < 16; ++e) {
      hv[e] = hW[(size_t)b * NH + lane * 16 + e];
      vv[e] = attn_v[lane * 16 + e];
    }
    float sc[4];
#pragma unroll
    for (int q = 0; q < 4; ++q) {
      sc[q] = -1e30f;
      for (int j = 0; j < 32; ++j) {
        int ss = q * 32 + j;
        const float* ep = EncProj + (size_t)(ss * NB + b) * NH + lane * 16;
        float part = 0.f;
#pragma unroll
        for (int e = 0; e < 16; ++e) part += tanhf(hv[e] + ep[e]) * vv[e];
#pragma unroll
        for (int off = 16; off > 0; off >>= 1) part += __shfl_xor(part, off, 32);
        float val = (src[b * NS + ss] != 1) ? part : -1e10f;
        if (j == lane) sc[q] = val;
      }
    }
    float m = fmaxf(fmaxf(sc[0], sc[1]), fmaxf(sc[2], sc[3]));
#pragma unroll
    for (int off = 16; off > 0; off >>= 1) m = fmaxf(m, __shfl_xor(m, off, 32));
    float sum = 0.f;
#pragma unroll
    for (int q = 0; q < 4; ++q) { sc[q] = expf(sc[q] - m); sum += sc[q]; }
#pragma unroll
    for (int off = 16; off > 0; off >>= 1) sum += __shfl_xor(sum, off, 32);
    float inv = 1.f / sum;
#pragma unroll
    for (int q = 0; q < 4; ++q) sc[q] *= inv;

    // ---- Phase C: weighted[b] = sum_s a[s] * EncOut[s,b,:] ----
    float wacc[16];
#pragma unroll
    for (int e = 0; e < 16; ++e) wacc[e] = 0.f;
#pragma unroll
    for (int q = 0; q < 4; ++q)
      for (int j = 0; j < 32; ++j) {
        int ss = q * 32 + j;
        float a = __shfl(sc[q], j, 32);
        const _Float16* eo = EncOut + (size_t)(ss * NB + b) * NH + lane * 16;
        v8h e0 = *reinterpret_cast<const v8h*>(eo);
        v8h e1 = *reinterpret_cast<const v8h*>(eo + 8);
#pragma unroll
        for (int e = 0; e < 8; ++e) {
          wacc[e]     += a * (float)e0[e];
          wacc[8 + e] += a * (float)e1[e];
        }
      }
#pragma unroll
    for (int e = 0; e < 16; ++e) {
      _Float16 hw = (_Float16)wacc[e];
      wbuf[b * NH + lane * 16 + e] = hw;
      Feat[(size_t)(t * NB + b) * 1536 + NH + lane * 16 + e] = hw;
    }
    __syncthreads();

    // ---- Phase D: gates = Gx[t] + weighted@Wih_w^T + h@Whh^T + biases ----
    v8f acc[4][2];
#pragma unroll
    for (int g = 0; g < 4; ++g)
#pragma unroll
      for (int jt = 0; jt < 2; ++jt) {
        int ccol = g * NH + jb + jt * 16 + cj;
        float bsum = bih[ccol] + bhh[ccol];
#pragma unroll
        for (int v = 0; v < 8; ++v)
          acc[g][jt][v] = Gx[(size_t)(t * NB + r0 + v) * 2048 + ccol] + bsum;
      }
    const _Float16* wpA = &wbuf[rr * NH + koff];
    const _Float16* hpA = &hbuf[rr * NH + koff];
    for (int k = 0; k < NH; k += 32) {
      v16h af = frag_ld(wpA + k);
#pragma unroll
      for (int g = 0; g < 4; ++g)
#pragma unroll
        for (int jt = 0; jt < 2; ++jt) {
          const _Float16* wp = Wih + (size_t)(g * NH + jb + jt * 16 + cj) * 1024 + NH + koff + k;
          acc[g][jt] = wmma_f16(af, frag_ld(wp), acc[g][jt]);
        }
    }
    for (int k = 0; k < NH; k += 32) {
      v16h af = frag_ld(hpA + k);
#pragma unroll
      for (int g = 0; g < 4; ++g)
#pragma unroll
        for (int jt = 0; jt < 2; ++jt) {
          const _Float16* wp = Whh + (size_t)(g * NH + jb + jt * 16 + cj) * NH + koff + k;
          acc[g][jt] = wmma_f16(af, frag_ld(wp), acc[g][jt]);
        }
    }
    __syncthreads();
#pragma unroll
    for (int jt = 0; jt < 2; ++jt)
#pragma unroll
      for (int v = 0; v < 8; ++v) {
        float si = sigm(acc[0][jt][v]);
        float sf = sigm(acc[1][jt][v]);
        float gg = tanhf(acc[2][jt][v]);
        float so = sigm(acc[3][jt][v]);
        float cc = sf * c[jt][v] + si * gg;
        c[jt][v] = cc;
        float hh = so * tanhf(cc);
        int row = r0 + v, jcol = jb + jt * 16 + cj;
        hbuf[row * NH + jcol] = (_Float16)hh;
        Feat[(size_t)(t * NB + row) * 1536 + jcol] = (_Float16)hh;
      }
    __syncthreads();
  }
}

// ---------------- small helper kernels ----------------
__global__ void cvt_f16(const float* __restrict__ s, _Float16* __restrict__ d, int n) {
  for (int i = blockIdx.x * blockDim.x + threadIdx.x; i < n; i += gridDim.x * blockDim.x)
    d[i] = (_Float16)s[i];
}

__global__ void embed_enc(const int* __restrict__ src, const float* __restrict__ emb,
                          _Float16* __restrict__ X) {
  int row = blockIdx.x;               // s*B + b (time-major)
  int s = row >> 5, b = row & 31;
  const float* e = emb + (size_t)src[b * NS + s] * NH;
  _Float16* x = X + (size_t)row * NH;
  for (int i = threadIdx.x; i < NH; i += blockDim.x) x[i] = (_Float16)e[i];
}

__global__ void embed_dec(const int* __restrict__ trg, const float* __restrict__ emb,
                          _Float16* __restrict__ Feat) {
  int row = blockIdx.x;               // t*B + b, t in 0..62
  int t = row >> 5, b = row & 31;
  const float* e = emb + (size_t)trg[b * NT + t] * NH;
  _Float16* x = Feat + (size_t)row * 1536 + 1024;
  for (int i = threadIdx.x; i < NH; i += blockDim.x) x[i] = (_Float16)e[i];
}

__global__ void tanh_vec(float* __restrict__ x, int n) {
  for (int i = blockIdx.x * blockDim.x + threadIdx.x; i < n; i += gridDim.x * blockDim.x)
    x[i] = tanhf(x[i]);
}

__global__ void zero_f32(float* __restrict__ x, int n) {
  for (int i = blockIdx.x * blockDim.x + threadIdx.x; i < n; i += gridDim.x * blockDim.x)
    x[i] = 0.f;
}

extern "C" void kernel_launch(void* const* d_in, const int* in_sizes, int n_in,
                              void* d_out, int out_size, void* d_ws, size_t ws_size,
                              hipStream_t stream) {
  const int*   src      = (const int*)  d_in[0];
  const int*   trg      = (const int*)  d_in[2];
  const float* enc_emb  = (const float*)d_in[3];
  const float* enc_wih  = (const float*)d_in[4];
  const float* enc_whh  = (const float*)d_in[5];
  const float* enc_bih  = (const float*)d_in[6];
  const float* enc_bhh  = (const float*)d_in[7];
  const float* enc_fc_w = (const float*)d_in[8];
  const float* enc_fc_b = (const float*)d_in[9];
  const float* dec_emb  = (const float*)d_in[10];
  const float* attn_w   = (const float*)d_in[11];
  const float* attn_b   = (const float*)d_in[12];
  const float* attn_v   = (const float*)d_in[13];
  const float* dec_wih  = (const float*)d_in[14];
  const float* dec_whh  = (const float*)d_in[15];
  const float* dec_bih  = (const float*)d_in[16];
  const float* dec_bhh  = (const float*)d_in[17];
  const float* out_w    = (const float*)d_in[18];
  const float* out_b    = (const float*)d_in[19];
  float* out = (float*)d_out;

  char* base = (char*)d_ws;
  size_t off = 0;
  auto alloc = [&](size_t bytes) {
    char* p = base + off;
    off += (bytes + 255) & ~(size_t)255;
    return p;
  };
  _Float16* wihE  = (_Float16*)alloc((size_t)2048 * 512 * 2);
  _Float16* whhE  = (_Float16*)alloc((size_t)2048 * 512 * 2);
  _Float16* wihD  = (_Float16*)alloc((size_t)2048 * 1024 * 2);
  _Float16* whhD  = (_Float16*)alloc((size_t)2048 * 512 * 2);
  _Float16* attW  = (_Float16*)alloc((size_t)512 * 1024 * 2);
  _Float16* fcW   = (_Float16*)alloc((size_t)512 * 512 * 2);
  _Float16* outW  = (_Float16*)alloc((size_t)NV * 1536 * 2);
  _Float16* Xenc  = (_Float16*)alloc((size_t)NS * NB * NH * 2);
  float*    GxE   = (float*)   alloc((size_t)NS * NB * 2048 * 4);
  _Float16* EncO  = (_Float16*)alloc((size_t)(NS * NB + 64) * NH * 2);   // +pad rows
  float*    EncP  = (float*)   alloc((size_t)NS * NB * NH * 4);
  float*    GxD   = (float*)   alloc((size_t)2048 * 2048 * 4);           // padded rows
  _Float16* Feat  = (_Float16*)alloc((size_t)2048 * 1536 * 2);           // padded rows
  float*    hid   = (float*)   alloc((size_t)NB * NH * 4);
  float*    cf    = (float*)   alloc((size_t)NB * NH * 4);
  float*    hWs   = (float*)   alloc((size_t)NB * NH * 4);

  // 1) weight conversions to f16
  cvt_f16<<<512, 256, 0, stream>>>(enc_wih, wihE, 2048 * 512);
  cvt_f16<<<512, 256, 0, stream>>>(enc_whh, whhE, 2048 * 512);
  cvt_f16<<<1024, 256, 0, stream>>>(dec_wih, wihD, 2048 * 1024);
  cvt_f16<<<512, 256, 0, stream>>>(dec_whh, whhD, 2048 * 512);
  cvt_f16<<<512, 256, 0, stream>>>(attn_w, attW, 512 * 1024);
  cvt_f16<<<256, 256, 0, stream>>>(enc_fc_w, fcW, 512 * 512);
  cvt_f16<<<4096, 256, 0, stream>>>(out_w, outW, NV * 1536);

  // 2) embeddings
  embed_enc<<<NS * NB, 256, 0, stream>>>(src, enc_emb, Xenc);
  embed_dec<<<(NT - 1) * NB, 256, 0, stream>>>(trg, dec_emb, Feat);

  // 3) encoder input-side gates for all steps: [4096,512] @ [512,2048]
  gemm_wmma<<<dim3(2048 / 128, 4096 / 64), 256, 0, stream>>>(
      Xenc, NH, wihE, NH, nullptr, GxE, 2048, 2048, NH, 4096);

  // 4) encoder recurrence (persistent single workgroup)
  enc_recurrent<<<1, 1024, 0, stream>>>(GxE, whhE, enc_bih, enc_bhh, EncO, cf);

  // 5) hidden = tanh(h_final @ fc_w^T + fc_b)   (M=32, padded A rows exist in EncO)
  gemm_wmma<<<dim3(512 / 128, 1), 256, 0, stream>>>(
      EncO + (size_t)(NS - 1) * NB * NH, NH, fcW, NH, enc_fc_b, hid, NH, NH, NH, NB);
  tanh_vec<<<64, 256, 0, stream>>>(hid, NB * NH);

  // 6) step-invariant attention term: enc @ We^T + attn_b
  gemm_wmma<<<dim3(512 / 128, 4096 / 64), 256, 0, stream>>>(
      EncO, NH, attW + NH, 1024, attn_b, EncP, NH, NH, NH, 4096);

  // 7) decoder embedding-side gates for all steps (Mpad=2048, M=2016)
  gemm_wmma<<<dim3(2048 / 128, 2048 / 64), 256, 0, stream>>>(
      Feat + 1024, 1536, wihD, 1024, nullptr, GxD, 2048, 2048, NH, (NT - 1) * NB);

  // 8) decoder recurrence (persistent single workgroup)
  dec_recurrent<<<1, 1024, 0, stream>>>(GxD, wihD, whhD, attW, dec_bih, dec_bhh,
                                        attn_v, src, hid, cf, EncO, EncP, hWs, Feat);

  // 9) outputs: row 0 zeros, then one big projection [2016,1536] @ [1536,32000]
  zero_f32<<<1024, 256, 0, stream>>>(out, NB * NV);
  gemm_wmma<<<dim3(NV / 128, 2048 / 64), 256, 0, stream>>>(
      Feat, 1536, outW, 1536, out_b, out + (size_t)NB * NV, NV, NV, 1536, (NT - 1) * NB);
}
